// FakeQuantLinear_64415919506225
// MI455X (gfx1250) — matrix operand order repaired and checked by reading, not compiled, not run on hardware
//
#include <hip/hip_runtime.h>

typedef _Float16 h16 __attribute__((ext_vector_type(16)));
typedef _Float16 h4  __attribute__((ext_vector_type(4)));
typedef float    f8  __attribute__((ext_vector_type(8)));

union Frag16 { h16 v; uint4 q[2]; };
union H4U    { h4 v; unsigned long long u; };

#define QMIN (-128.0f)
#define QMAX (127.0f)

#if __has_builtin(__builtin_amdgcn_global_load_async_to_lds_b128)
#define USE_ASYNC_LDS 1
#else
#define USE_ASYNC_LDS 0
#endif

#if USE_ASYNC_LDS
// builtin signature (from hipcc diagnostics):
//   arg0: native vector 'int __attribute__((vector_size(16)))' addrspace(1)* (global src)
//   arg1: LDS destination, addrspace(3)
//   arg2/arg3: immediate ints (offset, cpol)
typedef int vint4 __attribute__((vector_size(16)));
typedef __attribute__((address_space(1))) vint4 gvec4;
typedef __attribute__((address_space(3))) vint4 lvec4;
__device__ __forceinline__ gvec4* as_global(const void* p) {
  return (gvec4*)(unsigned long long)p;  // integer round-trip drops const, sets AS(1)
}
__device__ __forceinline__ lvec4* as_lds(void* p) {
  // generic LDS pointer low 32 bits == LDS byte offset (flat aperture rules)
  return (lvec4*)(unsigned int)(unsigned long long)p;
}
#if __has_builtin(__builtin_amdgcn_s_wait_asynccnt)
#define WAIT_ASYNC(n) __builtin_amdgcn_s_wait_asynccnt(n)
#else
#define WAIT_ASYNC(n) asm volatile("s_wait_asynccnt %0" ::"i"(n) : "memory")
#endif
#endif

// ------------------------------------------------------------------
// Prep 1: activation fake-quant  x -> f16 A[n,d]
//   A = (clip(rint(x / (r_s*scale_a)) + zp, -128, 127) - zp) * scale_a
// ------------------------------------------------------------------
__global__ __launch_bounds__(256) void fq_act_kernel(
    const float* __restrict__ x, const float* __restrict__ r_s,
    const float* __restrict__ scale_a, const int* __restrict__ zp_a,
    _Float16* __restrict__ a_h, int D, long total4) {
  long i4 = (long)blockIdx.x * 256 + threadIdx.x;
  if (i4 >= total4) return;
  long base = i4 * 4;
  int d = (int)(base % (long)D);

  float4 xv = *(const float4*)(x + base);
  float4 rv = *(const float4*)(r_s + d);
  float4 sv = *(const float4*)(scale_a + d);
  int4   zv = *(const int4*)(zp_a + d);

  H4U o;
  float q;
  q = fminf(fmaxf(rintf(xv.x / (rv.x * sv.x)) + (float)zv.x, QMIN), QMAX);
  o.v[0] = (_Float16)((q - (float)zv.x) * sv.x);
  q = fminf(fmaxf(rintf(xv.y / (rv.y * sv.y)) + (float)zv.y, QMIN), QMAX);
  o.v[1] = (_Float16)((q - (float)zv.y) * sv.y);
  q = fminf(fmaxf(rintf(xv.z / (rv.z * sv.z)) + (float)zv.z, QMIN), QMAX);
  o.v[2] = (_Float16)((q - (float)zv.z) * sv.z);
  q = fminf(fmaxf(rintf(xv.w / (rv.w * sv.w)) + (float)zv.w, QMIN), QMAX);
  o.v[3] = (_Float16)((q - (float)zv.w) * sv.w);

  *(unsigned long long*)(a_h + base) = o.u;
}

// ------------------------------------------------------------------
// Prep 2: weight dequant  -> f16 B[o,d] = (w_int - wzp[o]) * ws[o]
// ------------------------------------------------------------------
__global__ __launch_bounds__(256) void dq_w_kernel(
    const int* __restrict__ w_int, const float* __restrict__ w_scale,
    const int* __restrict__ w_zp, _Float16* __restrict__ b_h,
    int D, long total4) {
  long i4 = (long)blockIdx.x * 256 + threadIdx.x;
  if (i4 >= total4) return;
  long base = i4 * 4;
  int o = (int)(base / (long)D);

  int4 wv = *(const int4*)(w_int + base);
  float s = w_scale[o];
  float z = (float)w_zp[o];

  H4U out;
  out.v[0] = (_Float16)(((float)wv.x - z) * s);
  out.v[1] = (_Float16)(((float)wv.y - z) * s);
  out.v[2] = (_Float16)(((float)wv.z - z) * s);
  out.v[3] = (_Float16)(((float)wv.w - z) * s);
  *(unsigned long long*)(b_h + base) = out.u;
}

// ------------------------------------------------------------------
// GEMM: C[n,o] = sum_d A[n,d] * B[o,d] + bias[o]
//   128x128 block tile, BK=64 (2 WMMA k-steps), 8 wave32 waves,
//   each wave = 4x2 WMMA tiles -> 16 v_wmma per main-loop iteration.
//   Double-buffered LDS fed by GLOBAL_LOAD_ASYNC_TO_LDS_B128 (ASYNCcnt)
//   when available, register-staged b128 copies otherwise.
// ------------------------------------------------------------------
#define BM 128
#define BN 128
#define BK 64

__global__ __launch_bounds__(256) void gemm_wmma_kernel(
    const _Float16* __restrict__ A, const _Float16* __restrict__ B,
    const float* __restrict__ bias, float* __restrict__ C,
    int D, int O) {
  __shared__ __align__(16) _Float16 As[2][BM * BK];
  __shared__ __align__(16) _Float16 Bs[2][BN * BK];

  const int tid  = threadIdx.x;
  const int lane = tid & 31;
  const int wave = tid >> 5;
  const int wm   = wave & 1;   // 0..1 : 64-row half of block tile
  const int wn   = wave >> 1;  // 0..3 : 32-col slice
  const long RM  = (long)blockIdx.y * BM;
  const long RN  = (long)blockIdx.x * BN;
  const int  mr  = lane & 15;
  const bool hi  = lane >= 16;

  const int KT = D / BK;  // 32 iterations for D=2048

  // tile copy map: 1024 uint4 per matrix tile, 4 per thread each
  // u = tid + 256*j : row = u>>3 (64 f16 = 8 uint4 per row), kc = (u&7)*8
  int rowA[4], kcA[4];
#pragma unroll
  for (int j = 0; j < 4; ++j) {
    int u = tid + 256 * j;
    rowA[j] = u >> 3;
    kcA[j]  = (u & 7) * 8;
  }

  f8 acc[4][2];
#pragma unroll
  for (int mi = 0; mi < 4; ++mi)
#pragma unroll
    for (int ni = 0; ni < 2; ++ni)
#pragma unroll
      for (int r = 0; r < 8; ++r) acc[mi][ni][r] = 0.0f;

  const int akoff = hi ? 8 : 0;    // A frag: lanes16-31 start at K=8 within 32-chunk
  const int bkoff = hi ? 8 : 0;    // B frag: lanes16-31 start at K=8 within 32-chunk

#if USE_ASYNC_LDS
  // ---- async global->LDS pipeline ----
#define ISSUE_TILE(buf, kt)                                                          \
  {                                                                                  \
    const long ko = (long)(kt) * BK;                                                 \
    _Pragma("unroll") for (int j = 0; j < 4; ++j) {                                  \
      __builtin_amdgcn_global_load_async_to_lds_b128(                                \
          as_global(A + (RM + rowA[j]) * D + ko + kcA[j]),                           \
          as_lds(&As[buf][rowA[j] * BK + kcA[j]]), 0, 0);                            \
      __builtin_amdgcn_global_load_async_to_lds_b128(                                \
          as_global(B + (RN + rowA[j]) * D + ko + kcA[j]),                           \
          as_lds(&Bs[buf][rowA[j] * BK + kcA[j]]), 0, 0);                            \
    }                                                                                \
  }

  ISSUE_TILE(0, 0);

  for (int kt = 0; kt < KT; ++kt) {
    const int cur = kt & 1;
    const bool more = (kt + 1) < KT;

    if (more) {
      ISSUE_TILE(cur ^ 1, kt + 1);
      WAIT_ASYNC(8);   // tile `cur` complete (in-order); next tile stays in flight
    } else {
      WAIT_ASYNC(0);
    }
    __syncthreads();   // all waves' LDS writes for `cur` visible

    h16 afrag[4], bfrag[2];
#pragma unroll
    for (int ks = 0; ks < 2; ++ks) {   // two 32-deep k-steps per BK=64 tile
#pragma unroll
      for (int mi = 0; mi < 4; ++mi) {
        const _Float16* p = &As[cur][(wm * 64 + mi * 16 + mr) * BK + ks * 32];
        Frag16 t;
        t.q[0] = *(const uint4*)(p + akoff);
        t.q[1] = *(const uint4*)(p + 16 + akoff);
        afrag[mi] = t.v;
      }
#pragma unroll
      for (int ni = 0; ni < 2; ++ni) {
        const _Float16* p = &Bs[cur][(wn * 32 + ni * 16 + mr) * BK + ks * 32 + bkoff];
        Frag16 t;
        t.q[0] = *(const uint4*)(p);
        t.q[1] = *(const uint4*)(p + 16);
        bfrag[ni] = t.v;
      }
#pragma unroll
      for (int mi = 0; mi < 4; ++mi)
#pragma unroll
        for (int ni = 0; ni < 2; ++ni)
          acc[mi][ni] = __builtin_amdgcn_wmma_f32_16x16x32_f16(
              false, afrag[mi], false, bfrag[ni],
              (short)0, acc[mi][ni], false, false);
    }

    if (more) __syncthreads();  // no wave still reads buf `cur` when next issue starts
  }
#undef ISSUE_TILE

#else
  // ---- fallback: register-staged global->LDS ----
  {
#pragma unroll
    for (int j = 0; j < 4; ++j) {
      uint4 ra = *(const uint4*)(A + (RM + rowA[j]) * D + kcA[j]);
      uint4 rb = *(const uint4*)(B + (RN + rowA[j]) * D + kcA[j]);
      *(uint4*)&As[0][rowA[j] * BK + kcA[j]] = ra;
      *(uint4*)&Bs[0][rowA[j] * BK + kcA[j]] = rb;
    }
  }
  __syncthreads();

  for (int kt = 0; kt < KT; ++kt) {
    const int cur = kt & 1;
    const int nxt = cur ^ 1;
    const bool more = (kt + 1) < KT;

    uint4 na[4], nb[4];
    if (more) {
      const long ko = (long)(kt + 1) * BK;
#pragma unroll
      for (int j = 0; j < 4; ++j) {
        na[j] = *(const uint4*)(A + (RM + rowA[j]) * D + ko + kcA[j]);
        nb[j] = *(const uint4*)(B + (RN + rowA[j]) * D + ko + kcA[j]);
      }
    }

    h16 afrag[4], bfrag[2];
#pragma unroll
    for (int ks = 0; ks < 2; ++ks) {
#pragma unroll
      for (int mi = 0; mi < 4; ++mi) {
        const _Float16* p = &As[cur][(wm * 64 + mi * 16 + mr) * BK + ks * 32];
        Frag16 t;
        t.q[0] = *(const uint4*)(p + akoff);
        t.q[1] = *(const uint4*)(p + 16 + akoff);
        afrag[mi] = t.v;
      }
#pragma unroll
      for (int ni = 0; ni < 2; ++ni) {
        const _Float16* p = &Bs[cur][(wn * 32 + ni * 16 + mr) * BK + ks * 32 + bkoff];
        Frag16 t;
        t.q[0] = *(const uint4*)(p);
        t.q[1] = *(const uint4*)(p + 16);
        bfrag[ni] = t.v;
      }
#pragma unroll
      for (int mi = 0; mi < 4; ++mi)
#pragma unroll
        for (int ni = 0; ni < 2; ++ni)
          acc[mi][ni] = __builtin_amdgcn_wmma_f32_16x16x32_f16(
              false, afrag[mi], false, bfrag[ni],
              (short)0, acc[mi][ni], false, false);
    }

    if (more) {
      __syncthreads();
#pragma unroll
      for (int j = 0; j < 4; ++j) {
        *(uint4*)&As[nxt][rowA[j] * BK + kcA[j]] = na[j];
        *(uint4*)&Bs[nxt][rowA[j] * BK + kcA[j]] = nb[j];
      }
      __syncthreads();
    }
  }
#endif

  // epilogue: C/D layout -> row = tileRow + r + (lane>=16 ? 8 : 0), col = lane%16
  const long rbase = RM + wm * 64 + (hi ? 8 : 0);
#pragma unroll
  for (int ni = 0; ni < 2; ++ni) {
    const long col = RN + wn * 32 + ni * 16 + mr;
    const float bv = bias[col];
#pragma unroll
    for (int mi = 0; mi < 4; ++mi) {
#pragma unroll
      for (int r = 0; r < 8; ++r) {
        const long row = rbase + mi * 16 + r;
        C[row * (long)O + col] = acc[mi][ni][r] + bv;
      }
    }
  }
}

// ------------------------------------------------------------------
extern "C" void kernel_launch(void* const* d_in, const int* in_sizes, int n_in,
                              void* d_out, int out_size, void* d_ws, size_t ws_size,
                              hipStream_t stream) {
  const float* x       = (const float*)d_in[0];
  const float* r_s     = (const float*)d_in[1];
  const float* scale_a = (const float*)d_in[2];
  const int*   zp_a    = (const int*)d_in[3];
  const int*   w_int   = (const int*)d_in[4];
  const float* w_scale = (const float*)d_in[5];
  const int*   w_zp    = (const int*)d_in[6];
  const float* bias    = (const float*)d_in[7];
  float* out = (float*)d_out;

  const int D = in_sizes[1];                 // 2048
  const int N = in_sizes[0] / D;             // 32768
  const int O = in_sizes[5];                 // 2048

  _Float16* A_h = (_Float16*)d_ws;                       // N*D f16 (128 MB)
  _Float16* B_h = A_h + (size_t)N * (size_t)D;           // O*D f16 (8 MB)

  const long nx4 = (long)N * D / 4;
  fq_act_kernel<<<(unsigned)((nx4 + 255) / 256), 256, 0, stream>>>(
      x, r_s, scale_a, zp_a, A_h, D, nx4);

  const long nw4 = (long)O * D / 4;
  dq_w_kernel<<<(unsigned)((nw4 + 255) / 256), 256, 0, stream>>>(
      w_int, w_scale, w_zp, B_h, D, nw4);

  dim3 grid(O / BN, N / BM);
  gemm_wmma_kernel<<<grid, 256, 0, stream>>>(A_h, B_h, bias, out, D, O);
}
